// HyperConv_82600811036942
// MI455X (gfx1250) — matrix operand order repaired and checked by reading, not compiled, not run on hardware
//
#include <hip/hip_runtime.h>
#include <hip/hip_bf16.h>
#include <stddef.h>

// Problem constants (from reference)
#define BATCH   8
#define TLEN    65536
#define KCOND   1024          // conditioning steps
#define SLEN    64            // samples per step (T/K)
#define ZDIM    128
#define HID     32
#define CIN     64
#define COUT    64
#define KSIZE   2
#define JDIM    (CIN*KSIZE*COUT)   // 8192 generated weights per group
#define CK      (CIN*KSIZE)        // 128 = inner dim of main GEMM
#define NGRP    (BATCH*KCOND)      // 8192 groups

typedef __attribute__((ext_vector_type(16))) _Float16 v16h;
typedef __attribute__((ext_vector_type(8)))  float    v8f;
typedef __attribute__((ext_vector_type(4)))  float    v4f;

// ---- WMMA fragment loaders (CDNA5 16-bit layouts, wave32) ----
// A (16x32, M x K): lane l: m = l&15, h = l>>4; dword p: K = (p<4 ? 2p : 2p+8) + 8h
__device__ __forceinline__ v16h frag_a16(const _Float16* base, int stride_h,
                                         int row0, int koff, int lane) {
  int m = lane & 15, hh = lane >> 4;
  const _Float16* rp = base + (size_t)(row0 + m) * stride_h + koff;
  union { v16h v; unsigned u[8]; } t;
#pragma unroll
  for (int p = 0; p < 8; ++p) {
    int kk = (p < 4 ? 2 * p : 2 * p + 8) + 8 * hh;
    t.u[p] = *(const unsigned*)(rp + kk);
  }
  return t.v;
}
// B (32x16, K x N), stored N-major (row = column n, 'stride_h' halves per n):
// lane l: n = l&15, h = l>>4; dword p: K = (p<4 ? 2p : 2p+8) + 8h
__device__ __forceinline__ v16h frag_b16(const _Float16* base, int stride_h,
                                         int col0, int koff, int lane) {
  int n = lane & 15, hh = lane >> 4;
  const _Float16* rp = base + (size_t)(col0 + n) * stride_h + koff;
  union { v16h v; unsigned u[8]; } t;
#pragma unroll
  for (int p = 0; p < 8; ++p) {
    int kk = (p < 4 ? 2 * p : 2 * p + 8) + 8 * hh;
    t.u[p] = *(const unsigned*)(rp + kk);
  }
  return t.v;
}

// ---------------------------------------------------------------------------
// Kernel 1: hypernet hidden layers + generated bias.
//   Ht[n*32+m]   (f16) = relu(w1 z + b1)   (B-matrix layout for kernel 2)
//   biasbuf[n*64+o] (f32) = w4 relu(w3 z + b3) + b4
// ---------------------------------------------------------------------------
__global__ void __launch_bounds__(64)
k1_hyper(const float* __restrict__ z,
         const float* __restrict__ w1, const float* __restrict__ b1,
         const float* __restrict__ w3, const float* __restrict__ b3,
         const float* __restrict__ w4, const float* __restrict__ b4,
         _Float16* __restrict__ Ht, float* __restrict__ biasbuf) {
  int n = blockIdx.x;
  int b = n >> 10, k = n & (KCOND - 1);
  int tid = threadIdx.x;
  __shared__ float zs[ZDIM];
  __shared__ float h2s[HID];

  for (int i = tid; i < ZDIM; i += 64)
    zs[i] = z[((size_t)(b * ZDIM + i)) * KCOND + k];
  __syncthreads();

  if (tid < HID) {
    float acc = b1[tid];
#pragma unroll 8
    for (int i = 0; i < ZDIM; ++i) acc += w1[tid * ZDIM + i] * zs[i];
    Ht[(size_t)n * HID + tid] = (_Float16)fmaxf(acc, 0.f);
  } else {
    int t = tid - HID;
    float acc = b3[t];
#pragma unroll 8
    for (int i = 0; i < ZDIM; ++i) acc += w3[t * ZDIM + i] * zs[i];
    h2s[t] = fmaxf(acc, 0.f);
  }
  __syncthreads();

  float acc = b4[tid];
#pragma unroll
  for (int m = 0; m < HID; ++m) acc += w4[tid * HID + m] * h2s[m];
  biasbuf[(size_t)n * COUT + tid] = acc;
}

// ---------------------------------------------------------------------------
// Kernel 2: weight generation GEMM over permuted M-index jj = o*128 + c
// (original j = c*64 + o), so Wbuf[n] is laid out [o][c] — exactly the
// B-matrix (N-major) layout kernel 3 wants, enabling a straight async copy.
//   Wbuf[n][jj] = sum_m w2[j(jj)][m]*Ht[n][m] + b2[j(jj)]
//   M = JDIM (jj), N = NGRP (n), K = 32.  64x64 tile per 128-thread block.
// ---------------------------------------------------------------------------
__global__ void __launch_bounds__(128)
k2_wgen(const float* __restrict__ w2, const float* __restrict__ b2,
        const _Float16* __restrict__ Ht, _Float16* __restrict__ Wbuf) {
  int j0 = blockIdx.x * 64;   // jj tile base
  int n0 = blockIdx.y * 64;
  int tid = threadIdx.x, lane = tid & 31, wave = tid >> 5;
  __shared__ __align__(16) _Float16 w2s[64][40];   // A tile [jj][m], padded
  __shared__ float cs[64][72];                     // C stage [n][jj], padded
  __shared__ float b2s[64];

  for (int i = tid; i < 64 * HID; i += 128) {
    int jjl = i >> 5, m = i & 31;
    int jj = j0 + jjl;
    int j = (jj & 127) * 64 + (jj >> 7);           // original w2 row
    w2s[jjl][m] = (_Float16)w2[(size_t)j * HID + m];
  }
  if (tid < 64) {
    int jj = j0 + tid;
    b2s[tid] = b2[(jj & 127) * 64 + (jj >> 7)];
  }
  __syncthreads();

  int mt = wave;  // 4 waves -> 4 M-tiles
  v16h a = frag_a16(&w2s[0][0], 40, mt * 16, 0, lane);
#pragma unroll
  for (int nt = 0; nt < 4; ++nt) {
    v16h bb = frag_b16(Ht, HID, n0 + nt * 16, 0, lane);
    v8f c = {};
    c = __builtin_amdgcn_wmma_f32_16x16x32_f16(false, a, false, bb,
                                               (short)0, c, false, false);
    int nn = lane & 15, hh = lane >> 4;
#pragma unroll
    for (int r = 0; r < 8; ++r)
      cs[nt * 16 + nn][mt * 16 + r + 8 * hh] = c[r];
  }
  __syncthreads();

  for (int i = tid; i < 4096; i += 128) {
    int nl = i >> 6, jl = i & 63;
    Wbuf[(size_t)(n0 + nl) * JDIM + j0 + jl] = (_Float16)(cs[nl][jl] + b2s[jl]);
  }
}

// ---------------------------------------------------------------------------
// Kernel 3: main batched GEMM.  One block per group n = b*K + k (256 thr).
//   C[s][o] = sum_c X[s][c] * W[c][o] + bias[o]
//   X[s][c] = c<64 ? x[b,c,k*64+s] : x[b,c-64,k*64+s-1] (zero pad at t=-1)
//   Wbuf[n] is [o][c] f16 -> async-copied into padded LDS rows.
//   out[b,o,k*64+s] = C[s][o]
// ---------------------------------------------------------------------------
__global__ void __launch_bounds__(256)
k3_main(const float* __restrict__ x, const _Float16* __restrict__ Wbuf,
        const float* __restrict__ biasbuf, float* __restrict__ out) {
  int n = blockIdx.x;
  int b = n >> 10, k = n & (KCOND - 1);
  int tid = threadIdx.x, lane = tid & 31, wave = tid >> 5;
  __shared__ __align__(16) _Float16 xs[64][136];   // [s][c]
  __shared__ __align__(16) _Float16 wt[64][136];   // [o][c]  (B-matrix N-major)
  __shared__ float cs[64][68];                     // [s][o]
  __shared__ float bs[64];

  // --- Async-copy this group's 16KB weight tile into padded LDS rows. ---
  {
    const _Float16* wsrc = Wbuf + (size_t)n * JDIM;
    unsigned ldsbase = (unsigned)(size_t)(void*)&wt[0][0];
#pragma unroll
    for (int q = tid; q < 1024; q += 256) {        // 1024 x 16B chunks
      int o = q >> 4, cb = (q & 15) * 8;
      unsigned laddr = ldsbase + (unsigned)(o * 136 + cb) * 2u;
      unsigned long long ga = (unsigned long long)(size_t)(wsrc + q * 8);
      asm volatile("global_load_async_to_lds_b128 %0, %1, off"
                   :: "v"(laddr), "v"(ga) : "memory");
    }
  }

  // --- Stage x: load each value once (float4 along t, nontemporal), write
  //     both unfold copies; 64-element halo column for the shifted half. ---
  {
    const float* xb = x + (size_t)b * CIN * TLEN + (size_t)k * SLEN;
    for (int q = tid; q < 1024; q += 256) {        // 64 c x 16 chunks of 4 s
      int c = q >> 4, s0 = (q & 15) * 4;
      v4f v4 = __builtin_nontemporal_load((const v4f*)(xb + (size_t)c * TLEN + s0));
#pragma unroll
      for (int r = 0; r < 4; ++r) {
        int s = s0 + r;
        _Float16 h = (_Float16)v4[r];
        xs[s][c] = h;                              // direct tap
        if (s + 1 < SLEN) xs[s + 1][c + CIN] = h;  // shifted tap
      }
    }
    if (tid < CIN) {                               // halo: t = k*64 - 1
      float v = (k > 0) ? xb[(size_t)tid * TLEN - 1] : 0.f;
      xs[0][tid + CIN] = (_Float16)v;
    }
  }
  if (tid < 64) bs[tid] = biasbuf[(size_t)n * COUT + tid];

  asm volatile("s_wait_asynccnt 0x0" ::: "memory");
  __syncthreads();

  // 8 waves x 2 tiles = 16 C-tiles (4 mt x 4 nt); K-loop of 4 (128/32).
  int mt = wave >> 1;
  int ntb = (wave & 1) * 2;
  v8f acc0 = {}, acc1 = {};
#pragma unroll
  for (int ks = 0; ks < 4; ++ks) {
    v16h a  = frag_a16(&xs[0][0], 136, mt * 16, ks * 32, lane);
    v16h b0 = frag_b16(&wt[0][0], 136, ntb * 16, ks * 32, lane);
    v16h b1 = frag_b16(&wt[0][0], 136, (ntb + 1) * 16, ks * 32, lane);
    acc0 = __builtin_amdgcn_wmma_f32_16x16x32_f16(false, a, false, b0,
                                                  (short)0, acc0, false, false);
    acc1 = __builtin_amdgcn_wmma_f32_16x16x32_f16(false, a, false, b1,
                                                  (short)0, acc1, false, false);
  }
  int nn = lane & 15, hh = lane >> 4;
#pragma unroll
  for (int r = 0; r < 8; ++r) {
    int s = mt * 16 + r + 8 * hh;
    cs[s][ntb * 16 + nn]       = acc0[r];
    cs[s][(ntb + 1) * 16 + nn] = acc1[r];
  }
  __syncthreads();

  // Write out coalesced along t (nontemporal: streamed, never re-read).
  for (int i = tid; i < SLEN * COUT; i += 256) {
    int o = i >> 6, s = i & 63;
    float v = cs[s][o] + bs[o];
    __builtin_nontemporal_store(
        v, &out[((size_t)(b * COUT + o)) * TLEN + (size_t)k * SLEN + s]);
  }
}

// ---------------------------------------------------------------------------
extern "C" void kernel_launch(void* const* d_in, const int* in_sizes, int n_in,
                              void* d_out, int out_size, void* d_ws, size_t ws_size,
                              hipStream_t stream) {
  (void)in_sizes; (void)n_in; (void)out_size; (void)ws_size;
  const float* x  = (const float*)d_in[0];
  const float* z  = (const float*)d_in[1];
  const float* w1 = (const float*)d_in[2];
  const float* b1 = (const float*)d_in[3];
  const float* w2 = (const float*)d_in[4];
  const float* b2 = (const float*)d_in[5];
  const float* w3 = (const float*)d_in[6];
  const float* b3 = (const float*)d_in[7];
  const float* w4 = (const float*)d_in[8];
  const float* b4 = (const float*)d_in[9];

  // Workspace layout: Wbuf f16 [NGRP*JDIM] (128MB) | Ht f16 [NGRP*32] | bias f32 [NGRP*64]
  char* ws = (char*)d_ws;
  _Float16* Wbuf = (_Float16*)ws;
  _Float16* Ht   = (_Float16*)(ws + (size_t)NGRP * JDIM * sizeof(_Float16));
  float* biasbuf = (float*)(ws + (size_t)NGRP * JDIM * sizeof(_Float16)
                               + (size_t)NGRP * HID * sizeof(_Float16));

  k1_hyper<<<NGRP, 64, 0, stream>>>(z, w1, b1, w3, b3, w4, b4, Ht, biasbuf);
  k2_wgen<<<dim3(JDIM / 64, NGRP / 64), 128, 0, stream>>>(w2, b2, Ht, Wbuf);
  k3_main<<<NGRP, 256, 0, stream>>>(x, Wbuf, biasbuf, (float*)d_out);
}
